// BPPSModel_13151189860461
// MI455X (gfx1250) — compile-verified, matrix operand order checked by reading, not run on hardware
//
#include <hip/hip_runtime.h>
#include <hip/hip_bf16.h>

#define N_ATOMS   400000
#define NFEAT     512
#define HDIM      256
#define NTYPES    4
#define NSTRUCT   4096
#define ROWS      128          // atoms per workgroup tile
#define NTHREADS  512          // 16 wave32s
#define LN_EPS    1e-5f

typedef __bf16 v16bf __attribute__((ext_vector_type(16)));
typedef __bf16 v2bf  __attribute__((ext_vector_type(2)));
typedef float  v8f   __attribute__((ext_vector_type(8)));
typedef float  v16f  __attribute__((ext_vector_type(16)));

// ---------------------------------------------------------------------------
// init: zero output accumulator and per-type counters (graph is replayed, so
// this runs every launch).
// ---------------------------------------------------------------------------
__global__ void init_kernel(float* __restrict__ out, int n, int* __restrict__ cnt) {
    int i = blockIdx.x * blockDim.x + threadIdx.x;
    if (i < n) out[i] = 0.0f;
    if (i < NTYPES) cnt[i] = 0;
}

// ---------------------------------------------------------------------------
// pack: fp32 weights [T][K][256] -> bf16 in WMMA B-fragment order:
//   dst[(((t*Ktiles + kt)*16 + nt)*32 + lane)*16 + j]
//     = W[t][kt*32 + (lane>>4)*16 + j][nt*16 + (lane&15)]
// so each lane's 16 bf16 elements are one contiguous 32B vector load.
// ---------------------------------------------------------------------------
__global__ void pack_kernel(const float* __restrict__ W, __bf16* __restrict__ Wb,
                            int Ktiles, int K) {
    int id = blockIdx.x * blockDim.x + threadIdx.x;
    int total = NTYPES * Ktiles * 16 * 32;
    if (id >= total) return;
    int lane = id & 31;
    int nt   = (id >> 5) & 15;
    int kt   = (id >> 9) % Ktiles;
    int t    = id / (Ktiles * 512);
    int kbase = kt * 32 + (lane >> 4) * 16;
    int n     = nt * 16 + (lane & 15);
    size_t dst = (size_t)id * 16;
    #pragma unroll
    for (int j = 0; j < 16; ++j)
        Wb[dst + j] = (__bf16)W[((size_t)t * K + kbase + j) * HDIM + n];
}

// ---------------------------------------------------------------------------
// bucket: counting pass, builds 4 per-type atom-index lists.
// ---------------------------------------------------------------------------
__global__ void bucket_kernel(const int* __restrict__ numbers,
                              int* __restrict__ cnt, int* __restrict__ bucket) {
    int i = blockIdx.x * blockDim.x + threadIdx.x;
    if (i < N_ATOMS) {
        int t = numbers[i] & (NTYPES - 1);
        int p = atomicAdd(&cnt[t], 1);
        bucket[(size_t)t * N_ATOMS + p] = i;
    }
}

// ---------------------------------------------------------------------------
// fused MLP: 128 atoms of one type per workgroup, 16 wave32s.
// Wave w owns a 16-atom x 128-col stripe (8 f32 accumulators = 64 VGPRs ->
// accumulators stay pinned, WMMAs accumulate in place).
// GEMM1 (bf16 WMMA, K=512) -> LDS -> LN1+ReLU (4 threads/row, bf16 in place)
// -> GEMM2 (K=256, A fragments straight from LDS) -> LDS
// -> LN2+ReLU fused with Wout dot + composition -> one f32 atomic per atom.
// ---------------------------------------------------------------------------
__global__ void __launch_bounds__(NTHREADS)
mlp_kernel(const float* __restrict__ features,
           const __bf16* __restrict__ W1b,
           const __bf16* __restrict__ W2b,
           const float* __restrict__ Wout,
           const float* __restrict__ g1, const float* __restrict__ b1,
           const float* __restrict__ g2, const float* __restrict__ b2,
           const float* __restrict__ comp_w,
           const int* __restrict__ batch,
           const int* __restrict__ bucket,
           const int* __restrict__ cnt,
           float* __restrict__ out)
{
    extern __shared__ char smem[];
    float* Hs    = (float*)smem;                              // [128][256] fp32
    int*   aidx  = (int*)(smem + ROWS * HDIM * 4);            // [128]
    float* lnred = (float*)(smem + ROWS * HDIM * 4 + 512);    // [512]

    const int t     = blockIdx.y;
    const int count = cnt[t];
    const int base  = blockIdx.x * ROWS;
    if (base >= count) return;

    const int tid     = threadIdx.x;
    const int wave    = tid >> 5;           // 0..15
    const int lane    = tid & 31;
    const int rowbase = (wave >> 1) * 16;   // 8 row-tiles of 16 atoms
    const int colbase = (wave & 1) * 8;     // which 8 of the 16 col-tiles
    const int mrow    = lane & 15;
    const int khalf   = (lane >> 4) * 16;   // K-half selector for fragments

    if (tid < ROWS) {
        int g = base + tid;
        aidx[tid] = bucket[(size_t)t * N_ATOMS + (g < count ? g : base)];
    }
    __syncthreads();

    const v8f vzero = {0.f, 0.f, 0.f, 0.f, 0.f, 0.f, 0.f, 0.f};

    // ---------------- GEMM1: H1[128x256] = X[128x512] @ W1[t] ----------------
    v8f acc[8];
    #pragma unroll
    for (int nt = 0; nt < 8; ++nt) acc[nt] = vzero;

    const float* arow = features + (size_t)aidx[rowbase + mrow] * NFEAT + khalf;
    const __bf16* w1t = W1b + (size_t)t * (NFEAT * HDIM) + (size_t)colbase * 512 + lane * 16;

    #pragma unroll
    for (int kt = 0; kt < 16; ++kt) {
        v16f af = *(const v16f*)(arow + kt * 32);
        if (kt < 15) __builtin_prefetch(arow + (kt + 1) * 32, 0, 3);  // WGP-scope
        v16bf a;
        #pragma unroll
        for (int j = 0; j < 16; ++j) a[j] = (__bf16)af[j];

        const __bf16* bbase = w1t + (size_t)kt * 8192;
        #pragma unroll
        for (int nt = 0; nt < 8; ++nt) {
            v16bf b = *(const v16bf*)(bbase + nt * 512);
            acc[nt] = __builtin_amdgcn_wmma_f32_16x16x32_bf16(
                false, a, false, b, (short)0, acc[nt], false, false);
        }
    }

    // scatter accumulators to LDS (C layout: lane -> col, vgpr r -> row)
    #pragma unroll
    for (int nt = 0; nt < 8; ++nt) {
        #pragma unroll
        for (int r = 0; r < 8; ++r)
            Hs[(rowbase + (lane >> 4) * 8 + r) * HDIM + (colbase + nt) * 16 + mrow]
                = acc[nt][r];
    }
    __syncthreads();

    // -------- LayerNorm1 + ReLU: 4 threads/row, rewrite row as bf16 ---------
    {
        const int row = tid >> 2;
        const int q   = tid & 3;
        const int j0  = q * 64;
        float* hr = Hs + row * HDIM;

        float s = 0.f;
        for (int j = 0; j < 64; ++j) s += hr[j0 + j];
        lnred[tid] = s;
        __syncthreads();
        float mu = (lnred[(row << 2) + 0] + lnred[(row << 2) + 1] +
                    lnred[(row << 2) + 2] + lnred[(row << 2) + 3]) * (1.0f / HDIM);
        __syncthreads();

        float v = 0.f;
        for (int j = 0; j < 64; ++j) { float d = hr[j0 + j] - mu; v += d * d; }
        lnred[tid] = v;
        __syncthreads();
        float var = (lnred[(row << 2) + 0] + lnred[(row << 2) + 1] +
                     lnred[(row << 2) + 2] + lnred[(row << 2) + 3]) * (1.0f / HDIM);
        float inv = rsqrtf(var + LN_EPS);

        v2bf pk[32];                       // normalized chunk, packed in VGPRs
        #pragma unroll
        for (int jj = 0; jj < 32; ++jj) {
            int j = j0 + 2 * jj;
            float v0 = (hr[j]     - mu) * inv * g1[j]     + b1[j];
            float v1 = (hr[j + 1] - mu) * inv * g1[j + 1] + b1[j + 1];
            v2bf p;
            p.x = (__bf16)fmaxf(v0, 0.f);
            p.y = (__bf16)fmaxf(v1, 0.f);
            pk[jj] = p;
        }
        __syncthreads();                   // all fp32 reads done before bf16 writes
        v2bf* dst = (v2bf*)((__bf16*)hr + j0);
        #pragma unroll
        for (int jj = 0; jj < 32; ++jj) dst[jj] = pk[jj];
    }
    __syncthreads();

    // ---------------- GEMM2: H2[128x256] = H1b @ W2[t], A from LDS ----------
    v8f acc2[8];
    #pragma unroll
    for (int nt = 0; nt < 8; ++nt) acc2[nt] = vzero;

    const __bf16* arow2 = (const __bf16*)(Hs + (rowbase + mrow) * HDIM) + khalf;
    const __bf16* w2t = W2b + (size_t)t * (HDIM * HDIM) + (size_t)colbase * 512 + lane * 16;

    #pragma unroll
    for (int kt = 0; kt < 8; ++kt) {
        v16bf a = *(const v16bf*)(arow2 + kt * 32);      // 32B/lane ds load
        const __bf16* bbase = w2t + (size_t)kt * 8192;
        #pragma unroll
        for (int nt = 0; nt < 8; ++nt) {
            v16bf b = *(const v16bf*)(bbase + nt * 512);
            acc2[nt] = __builtin_amdgcn_wmma_f32_16x16x32_bf16(
                false, a, false, b, (short)0, acc2[nt], false, false);
        }
    }
    __syncthreads();   // everyone done reading H1b before overwrite

    #pragma unroll
    for (int nt = 0; nt < 8; ++nt) {
        #pragma unroll
        for (int r = 0; r < 8; ++r)
            Hs[(rowbase + (lane >> 4) * 8 + r) * HDIM + (colbase + nt) * 16 + mrow]
                = acc2[nt][r];
    }
    __syncthreads();

    // --- LayerNorm2 + ReLU fused with Wout dot + composition, 4 thr/row -----
    {
        const int row = tid >> 2;
        const int q   = tid & 3;
        const int j0  = q * 64;
        const float* hr = Hs + row * HDIM;

        float s = 0.f;
        for (int j = 0; j < 64; ++j) s += hr[j0 + j];
        lnred[tid] = s;
        __syncthreads();
        float mu = (lnred[(row << 2) + 0] + lnred[(row << 2) + 1] +
                    lnred[(row << 2) + 2] + lnred[(row << 2) + 3]) * (1.0f / HDIM);
        __syncthreads();

        float v = 0.f;
        for (int j = 0; j < 64; ++j) { float d = hr[j0 + j] - mu; v += d * d; }
        lnred[tid] = v;
        __syncthreads();
        float var = (lnred[(row << 2) + 0] + lnred[(row << 2) + 1] +
                     lnred[(row << 2) + 2] + lnred[(row << 2) + 3]) * (1.0f / HDIM);
        float inv = rsqrtf(var + LN_EPS);

        const float* wo = Wout + (size_t)t * HDIM;
        float e = 0.f;
        for (int j = 0; j < 64; ++j) {
            int jj = j0 + j;
            float x = (hr[jj] - mu) * inv * g2[jj] + b2[jj];
            e += fmaxf(x, 0.f) * wo[jj];
        }
        __syncthreads();
        lnred[tid] = e;
        __syncthreads();

        if (q == 0) {
            int g = base + row;
            if (g < count) {
                float etot = lnred[(row << 2) + 0] + lnred[(row << 2) + 1] +
                             lnred[(row << 2) + 2] + lnred[(row << 2) + 3] +
                             comp_w[t];
                atomicAdd(&out[batch[aidx[row]]], etot);
            }
        }
    }
}

// ---------------------------------------------------------------------------
extern "C" void kernel_launch(void* const* d_in, const int* in_sizes, int n_in,
                              void* d_out, int out_size, void* d_ws, size_t ws_size,
                              hipStream_t stream) {
    const float* features = (const float*)d_in[0];
    const float* W1       = (const float*)d_in[1];
    const float* W2       = (const float*)d_in[2];
    const float* Wout     = (const float*)d_in[3];
    const float* g1       = (const float*)d_in[4];
    const float* b1       = (const float*)d_in[5];
    const float* g2       = (const float*)d_in[6];
    const float* b2       = (const float*)d_in[7];
    const float* comp_w   = (const float*)d_in[8];
    const int*   numbers  = (const int*)d_in[9];
    const int*   batch    = (const int*)d_in[10];
    float*       out      = (float*)d_out;

    char* ws = (char*)d_ws;
    int*    cnt    = (int*)ws;                                        // 4 ints (+pad)
    int*    bucket = (int*)(ws + 256);                                // 4*N_ATOMS ints
    __bf16* W1b    = (__bf16*)(ws + 256 + (size_t)NTYPES * N_ATOMS * 4);
    __bf16* W2b    = W1b + (size_t)NTYPES * NFEAT * HDIM;

    init_kernel<<<(out_size + 255) / 256, 256, 0, stream>>>(out, out_size, cnt);

    pack_kernel<<<(NTYPES * 16 * 512 + 255) / 256, 256, 0, stream>>>(W1, W1b, 16, NFEAT);
    pack_kernel<<<(NTYPES * 8 * 512 + 255) / 256, 256, 0, stream>>>(W2, W2b, 8, HDIM);

    bucket_kernel<<<(N_ATOMS + 255) / 256, 256, 0, stream>>>(numbers, cnt, bucket);

    dim3 grid((N_ATOMS + ROWS - 1) / ROWS, NTYPES);
    // 128KB H tile + 512B indices + 2KB LN reduction scratch
    size_t shmem = (size_t)ROWS * HDIM * 4 + 512 + NTHREADS * 4;
    mlp_kernel<<<grid, NTHREADS, shmem, stream>>>(features, W1b, W2b, Wout,
                                                  g1, b1, g2, b2, comp_w,
                                                  batch, bucket, cnt, out);
}
